// FunnelRelMultiheadAttention_74431783240137
// MI455X (gfx1250) — matrix operand order repaired and checked by reading, not compiled, not run on hardware
//
#include <hip/hip_runtime.h>
#include <hip/hip_bf16.h>
#include <math.h>

// Problem constants (B=1)
#define S_LEN   2048
#define C_LEN   2048
#define DMODEL  768
#define NHEADS  12
#define DHEAD   64
#define QSCALE  0.125f   // 1/sqrt(64)

typedef __attribute__((ext_vector_type(16))) __bf16       bf16x16;
typedef __attribute__((ext_vector_type(8)))  float        f32x8;
typedef __attribute__((ext_vector_type(4)))  unsigned int u32x4;

union FragAB { u32x4 q[2]; bf16x16 v; };

__device__ __forceinline__ f32x8 zero8() {
  f32x8 z = {0.f,0.f,0.f,0.f,0.f,0.f,0.f,0.f};
  return z;
}

__device__ __forceinline__ unsigned short f2bf(float f) {
  union { float f; unsigned int u; } c; c.f = f;
  unsigned int r = c.u + 0x7FFFu + ((c.u >> 16) & 1u); // RNE
  return (unsigned short)(r >> 16);
}

// ---- WMMA fragment loaders (wave32 CDNA5 layouts, 16x16x32 bf16) ----
// A (16xK tile, row-major, ld elems): lane m=lane%16, half=lane/16.
//   values j=0..7  -> K = k0 + 8*half + j        (contiguous 16B)
//   values j=8..15 -> K = k0 + 16 + 8*half + j-8 (contiguous 16B)
__device__ __forceinline__ bf16x16 load_frag_a(const unsigned short* base, int ld,
                                               int k0, int lane) {
  int m = lane & 15, h = (lane >> 4) & 1;
  const unsigned short* p = base + (size_t)m * ld + k0 + h * 8;
  FragAB f;
  f.q[0] = *(const u32x4*)p;
  f.q[1] = *(const u32x4*)(p + 16);
  return f.v;
}
// B given as Bt (NxK tile, row-major, ld elems): lane n=lane%16, half=lane/16.
//   values j=0..7  -> K = k0 + 8*half + j; j=8..15 -> K = k0 + 16 + 8*half + j-8
__device__ __forceinline__ bf16x16 load_frag_b(const unsigned short* base, int ld,
                                               int k0, int lane) {
  int n = lane & 15, h = (lane >> 4) & 1;
  const unsigned short* p = base + (size_t)n * ld + k0 + h * 8;
  FragAB f;
  f.q[0] = *(const u32x4*)p;
  f.q[1] = *(const u32x4*)(p + 16);
  return f.v;
}

#define WMMA_BF16(A,B,Cacc) \
  __builtin_amdgcn_wmma_f32_16x16x32_bf16(false,(A),false,(B),(short)0,(Cacc),false,false)

// ---- CDNA5 async global->LDS copy (16B per lane, ASYNCcnt-tracked) ----
__device__ __forceinline__ void async_copy_b128(const void* gptr, void* lptr) {
  unsigned long long ga = (unsigned long long)(uintptr_t)gptr;
  unsigned int la = (unsigned int)(uintptr_t)lptr;   // low 32 bits = LDS offset
  asm volatile("global_load_async_to_lds_b128 %0, %1, off"
               :: "v"(la), "v"(ga) : "memory");
}
__device__ __forceinline__ void wait_asynccnt_le2() {
  asm volatile("s_wait_asynccnt 2" ::: "memory");
}
__device__ __forceinline__ void wait_asynccnt_0() {
  asm volatile("s_wait_asynccnt 0" ::: "memory");
}

// ---------------- elementwise helpers ----------------
__global__ void __launch_bounds__(256) cast_bf16_kernel(const float* __restrict__ src,
                                                        unsigned short* __restrict__ dst, int n) {
  int i = blockIdx.x * blockDim.x + threadIdx.x;
  if (i < n) dst[i] = f2bf(src[i]);
}

// src [R][C] f32 -> dst [C][R] bf16
__global__ void __launch_bounds__(256) transpose_cast_kernel(const float* __restrict__ src,
                                                             unsigned short* __restrict__ dst,
                                                             int R, int C) {
  int i = blockIdx.x * blockDim.x + threadIdx.x;
  if (i >= R * C) return;
  int r = i / C, c = i % C;
  dst[(size_t)c * R + r] = f2bf(src[i]);
}

// dst[i*768+c] = bf16(src[i*768+c] + bvec[c]*s)
__global__ void __launch_bounds__(256) addbias_cast_kernel(const float* __restrict__ src,
                                                           const float* __restrict__ bvec, float s,
                                                           unsigned short* __restrict__ dst, int n) {
  int i = blockIdx.x * blockDim.x + threadIdx.x;
  if (i >= n) return;
  int c = i % DMODEL;
  dst[i] = f2bf(src[i] + bvec[c] * s);
}

// tt_bias[n][i][0]=diff, [1]=same  (tiny: 12*2048 dot products of length 64)
__global__ void __launch_bounds__(256) ttbias_kernel(const float* __restrict__ qf,
                                                     const float* __restrict__ rsb,
                                                     const float* __restrict__ seg,
                                                     float* __restrict__ ttb) {
  int idx = blockIdx.x * blockDim.x + threadIdx.x;
  if (idx >= NHEADS * S_LEN) return;
  int n = idx / S_LEN, i = idx % S_LEN;
  float d0 = 0.f, d1 = 0.f;
  const float* q = qf + (size_t)i * DMODEL + n * DHEAD;
  const float* b = rsb + n * DHEAD;
  const float* s0 = seg + n * DHEAD;            // seg_embed[0]
  const float* s1 = seg + DMODEL + n * DHEAD;   // seg_embed[1]
  for (int h = 0; h < DHEAD; ++h) {
    float qv = q[h] + b[h] * QSCALE;
    d0 += qv * s0[h];
    d1 += qv * s1[h];
  }
  ttb[(size_t)idx * 2 + 0] = d0;
  ttb[(size_t)idx * 2 + 1] = d1;
}

// ---------------- generic WMMA GEMM: C[M,N] = alpha*(A @ Bt^T) + bias ----------------
// A: [M][K] bf16 row-major (lda), Bt: [N][K] bf16 row-major (ldb), C fp32 (ldc).
// Block: 128 thr = 4 waves; block tile 64x64; wave computes 16 rows x 64 cols.
__global__ void __launch_bounds__(128) gemm_bf16_kernel(const unsigned short* __restrict__ A, int lda,
                                                        const unsigned short* __restrict__ Bt, int ldb,
                                                        float* __restrict__ C, int ldc,
                                                        const float* __restrict__ bias,
                                                        float alpha, int K) {
  int lane = threadIdx.x & 31, wave = threadIdx.x >> 5;
  int half = lane >> 4, nlo = lane & 15;
  int m0 = blockIdx.y * 64 + wave * 16;
  int n0 = blockIdx.x * 64;
  f32x8 acc0 = zero8(), acc1 = zero8(), acc2 = zero8(), acc3 = zero8();
  const unsigned short* arow = A + (size_t)m0 * lda;
  for (int k0 = 0; k0 < K; k0 += 32) {
    bf16x16 a = load_frag_a(arow, lda, k0, lane);
    acc0 = WMMA_BF16(a, load_frag_b(Bt + (size_t)(n0 +  0) * ldb, ldb, k0, lane), acc0);
    acc1 = WMMA_BF16(a, load_frag_b(Bt + (size_t)(n0 + 16) * ldb, ldb, k0, lane), acc1);
    acc2 = WMMA_BF16(a, load_frag_b(Bt + (size_t)(n0 + 32) * ldb, ldb, k0, lane), acc2);
    acc3 = WMMA_BF16(a, load_frag_b(Bt + (size_t)(n0 + 48) * ldb, ldb, k0, lane), acc3);
  }
  #pragma unroll
  for (int t = 0; t < 4; ++t) {
    int c = n0 + t * 16 + nlo;
    float bv = bias ? bias[c] : 0.f;
    const f32x8& acc = (t == 0) ? acc0 : (t == 1) ? acc1 : (t == 2) ? acc2 : acc3;
    #pragma unroll
    for (int r = 0; r < 8; ++r) {
      int m = m0 + r + half * 8;
      C[(size_t)m * ldc + c] = acc[r] * alpha + bv;
    }
  }
}

// ---------------- qr -> u,w fused kernel ----------------
// qr[n][i][d] = sum_h qrin[i][n*64+h] * rk[d][n*64+h]; u = qr*phi[i][d]; w = qr*pi[i][d] (bf16)
__global__ void __launch_bounds__(128) qruw_kernel(const unsigned short* __restrict__ qrin,
                                                   const unsigned short* __restrict__ rkB,
                                                   const float* __restrict__ phi,
                                                   const float* __restrict__ piM,
                                                   unsigned short* __restrict__ uB,
                                                   unsigned short* __restrict__ wB) {
  int lane = threadIdx.x & 31, wave = threadIdx.x >> 5;
  int half = lane >> 4, nlo = lane & 15;
  int nh = blockIdx.z;
  int i0 = blockIdx.y * 64 + wave * 16;
  int d0 = blockIdx.x * 64;
  f32x8 acc0 = zero8(), acc1 = zero8(), acc2 = zero8(), acc3 = zero8();
  const unsigned short* A = qrin + (size_t)i0 * DMODEL + nh * DHEAD;
  #pragma unroll
  for (int k0 = 0; k0 < DHEAD; k0 += 32) {
    bf16x16 a = load_frag_a(A, DMODEL, k0, lane);
    acc0 = WMMA_BF16(a, load_frag_b(rkB + (size_t)(d0 +  0) * DMODEL + nh * DHEAD, DMODEL, k0, lane), acc0);
    acc1 = WMMA_BF16(a, load_frag_b(rkB + (size_t)(d0 + 16) * DMODEL + nh * DHEAD, DMODEL, k0, lane), acc1);
    acc2 = WMMA_BF16(a, load_frag_b(rkB + (size_t)(d0 + 32) * DMODEL + nh * DHEAD, DMODEL, k0, lane), acc2);
    acc3 = WMMA_BF16(a, load_frag_b(rkB + (size_t)(d0 + 48) * DMODEL + nh * DHEAD, DMODEL, k0, lane), acc3);
  }
  #pragma unroll
  for (int t = 0; t < 4; ++t) {
    int d = d0 + t * 16 + nlo;
    const f32x8& acc = (t == 0) ? acc0 : (t == 1) ? acc1 : (t == 2) ? acc2 : acc3;
    #pragma unroll
    for (int r = 0; r < 8; ++r) {
      int i = i0 + r + half * 8;
      float qr = acc[r];
      size_t o = ((size_t)nh * S_LEN + i) * DMODEL + d;
      uB[o] = f2bf(qr * phi[(size_t)i * DMODEL + d]);
      wB[o] = f2bf(qr * piM[(size_t)i * DMODEL + d]);
    }
  }
}

// ---------------- fused attention (flash-style) ----------------
// grid (S/64, NHEADS), 128 thr = 4 waves; wave owns 16 query rows, streams j in 32-col blocks.
// K-tile and V-tile are staged into LDS once per block via async global->LDS copies;
// the V-tile copy overlaps the entire K=768 positional WMMA stream.
__global__ void __launch_bounds__(128) flash_kernel(const unsigned short* __restrict__ qw,
                                                    const unsigned short* __restrict__ uB,
                                                    const unsigned short* __restrict__ wB,
                                                    const unsigned short* __restrict__ kb,
                                                    const unsigned short* __restrict__ psiB,
                                                    const unsigned short* __restrict__ omegaB,
                                                    const unsigned short* __restrict__ vT,
                                                    const float* __restrict__ ttb,
                                                    const float* __restrict__ cls,
                                                    const unsigned char* __restrict__ ttm,
                                                    const int* __restrict__ amask,
                                                    unsigned short* __restrict__ attnB) {
  __shared__ unsigned short lds_k[32][72];     // K tile  [j-local][h], padded stride
  __shared__ unsigned short lds_v[64][40];     // V^T tile [h][j-local], padded stride
  __shared__ unsigned short lds_p[4][16][40];  // per-wave P-tile transpose buffer
  int tid  = threadIdx.x;
  int lane = tid & 31, wave = tid >> 5;
  int half = lane >> 4, nlo = lane & 15;
  int nh = blockIdx.y;
  int i0 = blockIdx.x * 64 + wave * 16;

  const unsigned short* qwA = qw + (size_t)i0 * DMODEL + nh * DHEAD;
  const unsigned short* uA  = uB + ((size_t)nh * S_LEN + i0) * DMODEL;
  const unsigned short* wA  = wB + ((size_t)nh * S_LEN + i0) * DMODEL;
  const unsigned short* vtB = vT + (size_t)nh * DHEAD * C_LEN;

  f32x8 o0 = zero8(), o1 = zero8(), o2 = zero8(), o3 = zero8();
  float rm[8], rl[8], ttd[8], tts[8];
  #pragma unroll
  for (int r = 0; r < 8; ++r) {
    int i = i0 + r + half * 8;
    rm[r] = -1e30f; rl[r] = 0.f;
    ttd[r] = ttb[((size_t)nh * S_LEN + i) * 2 + 0];
    tts[r] = ttb[((size_t)nh * S_LEN + i) * 2 + 1];
  }

  for (int j0 = 0; j0 < C_LEN; j0 += 32) {
    // ---- issue async stages: K tile (32x64 bf16), then V tile (64x32 bf16) ----
    // 256 16B chunks each; 128 threads x 2 chunks. ASYNCcnt per wave: 2 (K) + 2 (V).
    #pragma unroll
    for (int q = 0; q < 2; ++q) {
      int c = tid * 2 + q;
      int r = c >> 3, col = (c & 7) * 8;   // K tile: 8 chunks per row of 64
      async_copy_b128(kb + (size_t)(j0 + r) * DMODEL + nh * DHEAD + col, &lds_k[r][col]);
    }
    #pragma unroll
    for (int q = 0; q < 2; ++q) {
      int c = tid * 2 + q;
      int r = c >> 2, col = (c & 3) * 8;   // V tile: 4 chunks per row of 32
      async_copy_b128(vtB + (size_t)r * C_LEN + j0 + col, &lds_v[r][col]);
    }
    if (j0 + 32 < C_LEN) {
      __builtin_prefetch(psiB + (size_t)(j0 + 32) * DMODEL, 0, 1);
      __builtin_prefetch(omegaB + (size_t)(j0 + 32) * DMODEL, 0, 1);
    }
    // K tile done (async loads complete in order); V may still be in flight.
    wait_asynccnt_le2();
    __syncthreads();

    // content score (K = 64) from LDS
    f32x8 s0 = zero8(), s1 = zero8();
    #pragma unroll
    for (int kk = 0; kk < DHEAD; kk += 32) {
      bf16x16 a = load_frag_a(qwA, DMODEL, kk, lane);
      s0 = WMMA_BF16(a, load_frag_b(&lds_k[0][0],  72, kk, lane), s0);
      s1 = WMMA_BF16(a, load_frag_b(&lds_k[16][0], 72, kk, lane), s1);
    }
    // positional score (K = 768, two streams u@psi^T + w@omega^T), direct from global/L2
    f32x8 p0 = zero8(), p1 = zero8();
    for (int kk = 0; kk < DMODEL; kk += 32) {
      bf16x16 au = load_frag_a(uA, DMODEL, kk, lane);
      p0 = WMMA_BF16(au, load_frag_b(psiB + (size_t)j0 * DMODEL, DMODEL, kk, lane), p0);
      p1 = WMMA_BF16(au, load_frag_b(psiB + (size_t)(j0 + 16) * DMODEL, DMODEL, kk, lane), p1);
      bf16x16 aw = load_frag_a(wA, DMODEL, kk, lane);
      p0 = WMMA_BF16(aw, load_frag_b(omegaB + (size_t)j0 * DMODEL, DMODEL, kk, lane), p0);
      p1 = WMMA_BF16(aw, load_frag_b(omegaB + (size_t)(j0 + 16) * DMODEL, DMODEL, kk, lane), p1);
    }
    // combine + masks
    int jA = j0 + nlo, jB = j0 + 16 + nlo;
    float amA = 1e6f * (1.f - (float)amask[jA]);
    float amB = 1e6f * (1.f - (float)amask[jB]);
    float sc0[8], sc1[8];
    #pragma unroll
    for (int r = 0; r < 8; ++r) {
      int i = i0 + r + half * 8;
      float tt0 = ttm[(size_t)i * C_LEN + jA] ? tts[r] : ttd[r];
      float tt1 = ttm[(size_t)i * C_LEN + jB] ? tts[r] : ttd[r];
      float c0 = cls[(size_t)i * C_LEN + jA];
      float c1 = cls[(size_t)i * C_LEN + jB];
      sc0[r] = s0[r] + (p0[r] + tt0) * c0 - amA;
      sc1[r] = s1[r] + (p1[r] + tt1) * c1 - amB;
    }
    // online softmax (row reductions across the 16-lane column group)
    float pr0[8], pr1[8];
    #pragma unroll
    for (int r = 0; r < 8; ++r) {
      float mx = fmaxf(sc0[r], sc1[r]);
      #pragma unroll
      for (int d = 1; d < 16; d <<= 1) mx = fmaxf(mx, __shfl_xor(mx, d, 32));
      float mn = fmaxf(rm[r], mx);
      float f = __expf(rm[r] - mn);
      rm[r] = mn;
      pr0[r] = __expf(sc0[r] - mn);
      pr1[r] = __expf(sc1[r] - mn);
      float rs = pr0[r] + pr1[r];
      #pragma unroll
      for (int d = 1; d < 16; d <<= 1) rs += __shfl_xor(rs, d, 32);
      rl[r] = rl[r] * f + rs;
      o0[r] *= f; o1[r] *= f; o2[r] *= f; o3[r] *= f;
    }
    // V tile now needed: wait remaining async copies, make visible block-wide
    wait_asynccnt_0();
    __syncthreads();
    // P tile (C-layout) -> LDS -> A-layout fragment for P@V
    #pragma unroll
    for (int r = 0; r < 8; ++r) {
      int m = r + half * 8;
      lds_p[wave][m][nlo]      = f2bf(pr0[r]);
      lds_p[wave][m][16 + nlo] = f2bf(pr1[r]);
    }
    __builtin_amdgcn_wave_barrier();   // same-wave LDS ops are in-order; just block reordering
    bf16x16 pa = load_frag_a(&lds_p[wave][0][0], 40, 0, lane);
    __builtin_amdgcn_wave_barrier();
    o0 = WMMA_BF16(pa, load_frag_b(&lds_v[ 0][0], 40, 0, lane), o0);
    o1 = WMMA_BF16(pa, load_frag_b(&lds_v[16][0], 40, 0, lane), o1);
    o2 = WMMA_BF16(pa, load_frag_b(&lds_v[32][0], 40, 0, lane), o2);
    o3 = WMMA_BF16(pa, load_frag_b(&lds_v[48][0], 40, 0, lane), o3);
    // protect lds_k/lds_v from next iteration's async writes
    __syncthreads();
  }
  // epilogue: normalize, store attn_vec (bf16) in [i][n*64+h] layout
  #pragma unroll
  for (int r = 0; r < 8; ++r) {
    int i = i0 + r + half * 8;
    float inv = 1.f / rl[r];
    size_t base = (size_t)i * DMODEL + nh * DHEAD + nlo;
    attnB[base +  0] = f2bf(o0[r] * inv);
    attnB[base + 16] = f2bf(o1[r] * inv);
    attnB[base + 32] = f2bf(o2[r] * inv);
    attnB[base + 48] = f2bf(o3[r] * inv);
  }
}

// ---------------- residual + layernorm ----------------
__global__ void __launch_bounds__(256) ln_kernel(const float* __restrict__ attn_out,
                                                 const float* __restrict__ query,
                                                 const float* __restrict__ lns,
                                                 const float* __restrict__ lnb,
                                                 float* __restrict__ out) {
  __shared__ float red[256];
  int i = blockIdx.x, t = threadIdx.x;
  float x[3]; float s = 0.f;
  #pragma unroll
  for (int k = 0; k < 3; ++k) {
    int c = t + k * 256;
    x[k] = query[(size_t)i * DMODEL + c] + attn_out[(size_t)i * DMODEL + c];
    s += x[k];
  }
  red[t] = s; __syncthreads();
  for (int o = 128; o > 0; o >>= 1) { if (t < o) red[t] += red[t + o]; __syncthreads(); }
  float mu = red[0] * (1.f / DMODEL);
  __syncthreads();
  float v = 0.f;
  #pragma unroll
  for (int k = 0; k < 3; ++k) { float d = x[k] - mu; v += d * d; }
  red[t] = v; __syncthreads();
  for (int o = 128; o > 0; o >>= 1) { if (t < o) red[t] += red[t + o]; __syncthreads(); }
  float inv = rsqrtf(red[0] * (1.f / DMODEL) + 1e-9f);
  #pragma unroll
  for (int k = 0; k < 3; ++k) {
    int c = t + k * 256;
    out[(size_t)i * DMODEL + c] = (x[k] - mu) * inv * lns[c] + lnb[c];
  }
}

// ---------------- host launch ----------------
extern "C" void kernel_launch(void* const* d_in, const int* in_sizes, int n_in,
                              void* d_out, int out_size, void* d_ws, size_t ws_size,
                              hipStream_t stream) {
  (void)in_sizes; (void)n_in; (void)out_size; (void)ws_size;
  const float* query = (const float*)d_in[0];
  const float* key   = (const float*)d_in[1];
  const float* value = (const float*)d_in[2];
  const float* phi   = (const float*)d_in[3];
  const float* piM   = (const float*)d_in[4];
  const float* psi   = (const float*)d_in[5];
  const float* omega = (const float*)d_in[6];
  const float* cls   = (const float*)d_in[7];
  const unsigned char* ttm = (const unsigned char*)d_in[8];
  const int*   amask = (const int*)d_in[9];
  const float* Wq  = (const float*)d_in[10];
  const float* Wk  = (const float*)d_in[11];
  const float* bk  = (const float*)d_in[12];
  const float* Wv  = (const float*)d_in[13];
  const float* bv  = (const float*)d_in[14];
  const float* rwb = (const float*)d_in[15];
  const float* rrb = (const float*)d_in[16];
  const float* rker= (const float*)d_in[17];
  const float* rsb = (const float*)d_in[18];
  const float* seg = (const float*)d_in[19];
  const float* Wo  = (const float*)d_in[20];
  const float* bo  = (const float*)d_in[21];
  const float* lns = (const float*)d_in[22];
  const float* lnb = (const float*)d_in[23];
  float* out = (float*)d_out;

  const size_t NE  = (size_t)S_LEN * DMODEL;          // 1572864
  const size_t NW  = (size_t)DMODEL * DMODEL;         // 589824
  const size_t NUW = (size_t)NHEADS * S_LEN * DMODEL; // 18.9M

  char* p = (char*)d_ws;
  auto alloc = [&p](size_t bytes) -> void* {
    void* r = (void*)p; p += (bytes + 255) & ~(size_t)255; return r;
  };
  unsigned short* qIn   = (unsigned short*)alloc(NE * 2);
  unsigned short* kIn   = (unsigned short*)alloc(NE * 2);
  unsigned short* vIn   = (unsigned short*)alloc(NE * 2);
  unsigned short* WqT   = (unsigned short*)alloc(NW * 2);
  unsigned short* WkT   = (unsigned short*)alloc(NW * 2);
  unsigned short* WvT   = (unsigned short*)alloc(NW * 2);
  unsigned short* WoT   = (unsigned short*)alloc(NW * 2);
  unsigned short* psiB  = (unsigned short*)alloc(NE * 2);
  unsigned short* omegaB= (unsigned short*)alloc(NE * 2);
  unsigned short* rkB   = (unsigned short*)alloc(NW * 2);
  float*          qf    = (float*)alloc(NE * 4);
  float*          kf    = (float*)alloc(NE * 4);
  float*          vf    = (float*)alloc(NE * 4);
  unsigned short* qwB   = (unsigned short*)alloc(NE * 2);
  unsigned short* qrin  = (unsigned short*)alloc(NE * 2);
  unsigned short* kbf   = (unsigned short*)alloc(NE * 2);
  unsigned short* vT    = (unsigned short*)alloc(NE * 2);
  unsigned short* uB    = (unsigned short*)alloc(NUW * 2);
  unsigned short* wB    = (unsigned short*)alloc(NUW * 2);
  float*          ttb   = (float*)alloc((size_t)NHEADS * S_LEN * 2 * 4);
  unsigned short* attnB = (unsigned short*)alloc(NE * 2);
  float*          attnO = (float*)alloc(NE * 4);

  const int T = 256;
  dim3 gNE((unsigned)((NE + T - 1) / T)), gNW((unsigned)((NW + T - 1) / T));

  cast_bf16_kernel<<<gNE, T, 0, stream>>>(query, qIn, (int)NE);
  cast_bf16_kernel<<<gNE, T, 0, stream>>>(key,   kIn, (int)NE);
  cast_bf16_kernel<<<gNE, T, 0, stream>>>(value, vIn, (int)NE);
  cast_bf16_kernel<<<gNE, T, 0, stream>>>(psi,   psiB, (int)NE);
  cast_bf16_kernel<<<gNE, T, 0, stream>>>(omega, omegaB, (int)NE);
  cast_bf16_kernel<<<gNW, T, 0, stream>>>(rker,  rkB, (int)NW);
  transpose_cast_kernel<<<gNW, T, 0, stream>>>(Wq, WqT, DMODEL, DMODEL);
  transpose_cast_kernel<<<gNW, T, 0, stream>>>(Wk, WkT, DMODEL, DMODEL);
  transpose_cast_kernel<<<gNW, T, 0, stream>>>(Wv, WvT, DMODEL, DMODEL);
  transpose_cast_kernel<<<gNW, T, 0, stream>>>(Wo, WoT, DMODEL, DMODEL);

  dim3 gproj(DMODEL / 64, S_LEN / 64);
  gemm_bf16_kernel<<<gproj, 128, 0, stream>>>(qIn, DMODEL, WqT, DMODEL, qf, DMODEL, nullptr, QSCALE, DMODEL);
  gemm_bf16_kernel<<<gproj, 128, 0, stream>>>(kIn, DMODEL, WkT, DMODEL, kf, DMODEL, bk, 1.f, DMODEL);
  gemm_bf16_kernel<<<gproj, 128, 0, stream>>>(vIn, DMODEL, WvT, DMODEL, vf, DMODEL, bv, 1.f, DMODEL);

  addbias_cast_kernel<<<gNE, T, 0, stream>>>(qf, rwb, QSCALE, qwB, (int)NE);
  addbias_cast_kernel<<<gNE, T, 0, stream>>>(qf, rrb, QSCALE, qrin, (int)NE);
  cast_bf16_kernel<<<gNE, T, 0, stream>>>(kf, kbf, (int)NE);
  transpose_cast_kernel<<<gNE, T, 0, stream>>>(vf, vT, C_LEN, DMODEL);
  ttbias_kernel<<<(NHEADS * S_LEN + T - 1) / T, T, 0, stream>>>(qf, rsb, seg, ttb);

  dim3 gqr(DMODEL / 64, S_LEN / 64, NHEADS);
  qruw_kernel<<<gqr, 128, 0, stream>>>(qrin, rkB, phi, piM, uB, wB);

  dim3 gfl(S_LEN / 64, NHEADS);
  flash_kernel<<<gfl, 128, 0, stream>>>(qwB, uB, wB, kbf, psiB, omegaB, vT, ttb, cls, ttm, amask, attnB);

  gemm_bf16_kernel<<<gproj, 128, 0, stream>>>(attnB, DMODEL, WoT, DMODEL, attnO, DMODEL, bo, 1.f, DMODEL);
  ln_kernel<<<S_LEN, 256, 0, stream>>>(attnO, query, lns, lnb, out);
}
